// CKNLayer_6528350290537
// MI455X (gfx1250) — matrix operand order repaired and checked by reading, not compiled, not run on hardware
//
#include <hip/hip_runtime.h>

typedef __attribute__((ext_vector_type(16))) __bf16 v16bf;
typedef __attribute__((ext_vector_type(8)))  __bf16 v8bf;
typedef __attribute__((ext_vector_type(8)))  float  v8f;
typedef __attribute__((ext_vector_type(2)))  float  v2f;
typedef __attribute__((ext_vector_type(4)))  unsigned int u32x4;
typedef __attribute__((ext_vector_type(8)))  int i32x8;
typedef __attribute__((ext_vector_type(4)))  int i32x4;

#define INV_SIG2 (1.0f/(0.6f*0.6f))
#define KWW_REG  1e-3f
#define EPSN     1e-10f

// ---------------------------------------------------------------- TDM support
#if defined(__has_builtin)
#if __has_builtin(__builtin_amdgcn_tensor_load_to_lds) && __has_builtin(__builtin_amdgcn_s_wait_tensorcnt)
#define HAVE_TDM 1
#endif
#endif
#ifndef HAVE_TDM
#define HAVE_TDM 0
#endif
__device__ constexpr bool USE_TDM = (HAVE_TDM != 0);

// 2-D tile load via Tensor Data Mover. lds_addr = raw LDS byte address.
// dims/strides/tiles in elements; dsz_log: 0=1B,1=2B,2=4B.
// pad_interval: pad after 2^(v+1) DWORDs; pad_amount: (v+1) DWORDs.
__device__ __forceinline__ void tdm_load_2d(const void* gsrc, unsigned lds_addr,
                                            unsigned dim0, unsigned dim1,
                                            unsigned stride0,
                                            unsigned tile0, unsigned tile1,
                                            unsigned dsz_log,
                                            unsigned pad_interval, unsigned pad_amount) {
#if HAVE_TDM
    unsigned long long ga = (unsigned long long)(size_t)gsrc;
    u32x4 g0;
    g0[0] = 1u;                                        // count=1 (valid user D#)
    g0[1] = lds_addr;                                  // lds_addr
    g0[2] = (unsigned)(ga & 0xFFFFFFFFu);              // global_addr[31:0]
    g0[3] = (unsigned)((ga >> 32) & 0x01FFFFFFu)       // global_addr[56:32]
          | (2u << 30);                                // type = 2 ("image")
    i32x8 g1;
    g1[0] = (int)((dsz_log << 16) | (1u << 20) |       // data_size, pad_enable
                  (pad_interval << 22) | (pad_amount << 25));
    g1[1] = (int)(dim0 << 16);                         // tensor_dim0[15:0] @bit48
    g1[2] = (int)((dim0 >> 16) | (dim1 << 16));        // dim0 hi | tensor_dim1 lo
    g1[3] = (int)((dim1 >> 16) | (tile0 << 16));       // dim1 hi | tile_dim0
    g1[4] = (int)tile1;                                // tile_dim1 | tile_dim2=0
    g1[5] = (int)stride0;                              // tensor_dim0_stride lo32
    g1[6] = 0;                                         // stride hi | dim1_stride lo
    g1[7] = 0;
    i32x4 z4 = {0, 0, 0, 0};
#if __clang_major__ >= 23
    i32x8 z8 = {0, 0, 0, 0, 0, 0, 0, 0};
    __builtin_amdgcn_tensor_load_to_lds(g0, g1, z4, z4, z8, 0);
#else
    __builtin_amdgcn_tensor_load_to_lds(g0, g1, z4, z4, 0);
#endif
#endif
}
__device__ __forceinline__ void tdm_join() {
#if HAVE_TDM
    __builtin_amdgcn_s_wait_tensorcnt(0);
#endif
}
__device__ __forceinline__ unsigned lds_addr_of(const void* p) {
    return (unsigned)(size_t)p;   // low 32 bits of flat shared pointer = LDS byte address
}

// ---------------------------------------------------------------- prep kernels
__global__ void k_prep_f(const float* __restrict__ f, __bf16* __restrict__ fb,
                         float* __restrict__ s2) {
    int idx = blockIdx.x * 256 + threadIdx.x;          // (b,y,x) in [0,65536)
    int b = idx >> 10, p = idx & 1023;
    const float* fp = f + ((size_t)b << 17) + p;
    __bf16* fbp = fb + ((size_t)b << 17) + p;
    float sum = 0.f;
#pragma unroll 4
    for (int c = 0; c < 128; ++c) {
        float v = fp[c << 10];
        sum += v * v;
        fbp[c << 10] = (__bf16)v;
    }
    s2[idx] = sum;
}

__global__ void k_norms(const float* __restrict__ s2, float* __restrict__ norms) {
    int idx = blockIdx.x * 256 + threadIdx.x;
    int b = idx >> 10, p = idx & 1023, y = p >> 5, x = p & 31;
    const float* sp = s2 + (b << 10);
    float s = 0.f;
#pragma unroll
    for (int dy = -1; dy <= 1; ++dy)
#pragma unroll
        for (int dx = -1; dx <= 1; ++dx) {
            int yy = y + dy, xx = x + dx;
            if ((unsigned)yy < 32u && (unsigned)xx < 32u) s += sp[yy * 32 + xx];
        }
    norms[idx] = fmaxf(sqrtf(s), EPSN);
}

__global__ void k_prep_w(const float* __restrict__ Wf, __bf16* __restrict__ Wb,
                         float* sumsq, int* cnt, int* gen) {
    int idx = blockIdx.x * 256 + threadIdx.x;
    if (idx < 512 * 1152) Wb[idx] = (__bf16)Wf[idx];
    if (idx == 0) { *sumsq = 0.f; *cnt = 0; *gen = 0; }
}

// ---------------------------------------------------------------- f32 WMMA GEMM core
__device__ __forceinline__ void gbar(int* cnt, int* gen) {
    __threadfence();
    __syncthreads();
    if (threadIdx.x == 0) {
        int g = atomicAdd(gen, 0);
        if (atomicAdd(cnt, 1) == 15) {
            atomicExch(cnt, 0);
            __threadfence();
            atomicAdd(gen, 1);
        } else {
            while (atomicAdd(gen, 0) == g) __builtin_amdgcn_s_sleep(2);
        }
    }
    __syncthreads();
    __threadfence();
}

#define LDAS 36
#define LDBS 132
// 128x128 tile of C = A@B (BT: use B^T where B stored row-major [n][k]).
template <bool BT>
__device__ void mm128(const float* __restrict__ A, const float* __restrict__ B,
                      int lda, int ldb, int K, int row0, int col0,
                      float* As, float* Bs, v8f acc[4][2]) {
    const int t = threadIdx.x, lane = t & 31, w = t >> 5;
    const int h = lane >> 4, l15 = lane & 15;
    const int wm = w >> 2, wn = w & 3;
#pragma unroll
    for (int i = 0; i < 4; ++i)
#pragma unroll
        for (int j = 0; j < 2; ++j)
#pragma unroll
            for (int e = 0; e < 8; ++e) acc[i][j][e] = 0.f;

    for (int k0 = 0; k0 < K; k0 += 32) {
        __syncthreads();                 // previous compute done, LDS free
        if (USE_TDM) {
            if (w == 0) {
                // A tile: 128 rows x 32 f32, LDS pitch 36 f32 (pad 4 DW after 32 DW)
                tdm_load_2d(A + (size_t)row0 * lda + k0, lds_addr_of(As),
                            (unsigned)lda, 512u, (unsigned)lda, 32u, 128u, 2u, 4u, 3u);
                if (!BT)   // B tile: 32 rows x 128 f32, LDS pitch 132 (pad 4 DW after 128 DW)
                    tdm_load_2d(B + (size_t)k0 * ldb + col0, lds_addr_of(Bs),
                                (unsigned)ldb, 512u, (unsigned)ldb, 128u, 32u, 2u, 6u, 3u);
            }
        } else {
            {   // A tile per-thread
                int m = t >> 1, cc = (t & 1) * 16;
                const float4* src = (const float4*)(A + (size_t)(row0 + m) * lda + k0 + cc);
                float4* dst = (float4*)&As[m * LDAS + cc];
#pragma unroll
                for (int i = 0; i < 4; ++i) dst[i] = src[i];
            }
            if (!BT) {
                int k = t >> 3, nn = (t & 7) * 16;
                const float4* src = (const float4*)(B + (size_t)(k0 + k) * ldb + col0 + nn);
                float4* dst = (float4*)&Bs[k * LDBS + nn];
#pragma unroll
                for (int i = 0; i < 4; ++i) dst[i] = src[i];
            }
        }
        if (BT) {   // B^T scatter (gram only): all threads
            int n = t >> 1, cc = (t & 1) * 16;
            const float* src = B + (size_t)(col0 + n) * ldb + k0 + cc;
#pragma unroll
            for (int i = 0; i < 16; ++i) Bs[(cc + i) * LDBS + n] = src[i];
        }
        if (USE_TDM && w == 0) tdm_join();
        __syncthreads();
#pragma unroll
        for (int s = 0; s < 8; ++s) {
            int kk = 4 * s + 2 * h;
            v2f a[4], bb[2];
#pragma unroll
            for (int tm = 0; tm < 4; ++tm) {
                int m = 64 * wm + 16 * tm + l15;
                a[tm][0] = As[m * LDAS + kk];
                a[tm][1] = As[m * LDAS + kk + 1];
            }
#pragma unroll
            for (int tn = 0; tn < 2; ++tn) {
                int n = 32 * wn + 16 * tn + l15;
                bb[tn][0] = Bs[kk * LDBS + n];
                bb[tn][1] = Bs[(kk + 1) * LDBS + n];
            }
#pragma unroll
            for (int tm = 0; tm < 4; ++tm)
#pragma unroll
                for (int tn = 0; tn < 2; ++tn)
                    acc[tm][tn] = __builtin_amdgcn_wmma_f32_16x16x4_f32(
                        false, a[tm], false, bb[tn], (short)0, acc[tm][tn], false, false);
        }
    }
}

// Persistent: gram -> ||.||_F -> 20 Newton-Schulz iterations -> NrmB (bf16, transposed)
__global__ void __launch_bounds__(256) k_newton(
    const float* __restrict__ Wf, float* gram, float* Y0, float* Y1,
    float* Z0, float* Z1, float* T, __bf16* NrmB,
    float* sumsq, int* cnt, int* gen) {
    __shared__ float As[128 * LDAS];
    __shared__ float Bs[32 * LDBS];
    const int t = threadIdx.x, lane = t & 31, w = t >> 5;
    const int h = lane >> 4, l15 = lane & 15;
    const int wm = w >> 2, wn = w & 3;
    const int row0 = (blockIdx.x >> 2) * 128, col0 = (blockIdx.x & 3) * 128;
    v8f acc[4][2];

    mm128<true>(Wf, Wf, 1152, 1152, 1152, row0, col0, As, Bs, acc);
    float part = 0.f;
#pragma unroll
    for (int tm = 0; tm < 4; ++tm)
#pragma unroll
        for (int tn = 0; tn < 2; ++tn)
#pragma unroll
            for (int r = 0; r < 8; ++r) {
                int M = row0 + 64 * wm + 16 * tm + r + 8 * h;
                int N = col0 + 32 * wn + 16 * tn + l15;
                float v = expf((acc[tm][tn][r] - 1.0f) * INV_SIG2);
                if (M == N) v += KWW_REG;
                gram[M * 512 + N] = v;
                part += v * v;
            }
    atomicAdd(sumsq, part);
    gbar(cnt, gen);

    float normA = sqrtf(*(volatile float*)sumsq);
    float invNA = 1.0f / normA;
    for (int i = t; i < 128 * 128; i += 256) {
        int M = row0 + (i >> 7), N = col0 + (i & 127);
        Y0[M * 512 + N] = gram[M * 512 + N] * invNA;
        Z0[M * 512 + N] = (M == N) ? 1.0f : 0.0f;
    }
    gbar(cnt, gen);

    float* Yp[2] = {Y0, Y1};
    float* Zp[2] = {Z0, Z1};
    for (int it = 0; it < 20; ++it) {
        int p = it & 1;
        mm128<false>(Zp[p], Yp[p], 512, 512, 512, row0, col0, As, Bs, acc);
#pragma unroll
        for (int tm = 0; tm < 4; ++tm)
#pragma unroll
            for (int tn = 0; tn < 2; ++tn)
#pragma unroll
                for (int r = 0; r < 8; ++r) {
                    int M = row0 + 64 * wm + 16 * tm + r + 8 * h;
                    int N = col0 + 32 * wn + 16 * tn + l15;
                    float v = -0.5f * acc[tm][tn][r];
                    if (M == N) v += 1.5f;
                    T[M * 512 + N] = v;
                }
        gbar(cnt, gen);
        if (it < 19) {
            mm128<false>(Yp[p], T, 512, 512, 512, row0, col0, As, Bs, acc);
#pragma unroll
            for (int tm = 0; tm < 4; ++tm)
#pragma unroll
                for (int tn = 0; tn < 2; ++tn)
#pragma unroll
                    for (int r = 0; r < 8; ++r) {
                        int M = row0 + 64 * wm + 16 * tm + r + 8 * h;
                        int N = col0 + 32 * wn + 16 * tn + l15;
                        Yp[1 - p][M * 512 + N] = acc[tm][tn][r];
                    }
        }
        mm128<false>(T, Zp[p], 512, 512, 512, row0, col0, As, Bs, acc);
#pragma unroll
        for (int tm = 0; tm < 4; ++tm)
#pragma unroll
            for (int tn = 0; tn < 2; ++tn)
#pragma unroll
                for (int r = 0; r < 8; ++r) {
                    int M = row0 + 64 * wm + 16 * tm + r + 8 * h;
                    int N = col0 + 32 * wn + 16 * tn + l15;
                    Zp[1 - p][M * 512 + N] = acc[tm][tn][r];
                }
        gbar(cnt, gen);
    }
    float scale = rsqrtf(normA);
    for (int i = t; i < 128 * 128; i += 256) {
        int M = row0 + (i >> 7), N = col0 + (i & 127);
        NrmB[N * 512 + M] = (__bf16)(Z0[M * 512 + N] * scale);
    }
}

// ---------------------------------------------------------------- stage A (bf16 WMMA)
__global__ void __launch_bounds__(256) k_stageA(
    const __bf16* __restrict__ fb, const __bf16* __restrict__ Wb,
    const float* __restrict__ norms, __bf16* __restrict__ PooledB) {
    __shared__ __bf16 As[2][128 * 40];
    __shared__ __bf16 Bs[2][256 * 40];
    __shared__ float normsS[128], invS[128];
    const int t = threadIdx.x, lane = t & 31, w = t >> 5;
    const int h = lane >> 4, l15 = lane & 15;
    const int wm = w >> 2, wn = w & 3;
    const int mt = blockIdx.x, nt = blockIdx.y;
    const int b = mt >> 3, y0 = (mt & 7) * 4;
    const int n0 = nt * 256;
    if (t < 128) {
        float nv = norms[mt * 128 + t];
        normsS[t] = nv;
        invS[t] = 1.0f / nv;
    }
    const __bf16* fbase = fb + ((size_t)b << 17);

    auto gatherA = [&](int buf, int k0) {     // fused im2col, per-thread scatter
        int m = t >> 1, ks = (t & 1) * 16;
        int y = y0 + (m >> 5), x = m & 31;
#pragma unroll
        for (int i = 0; i < 16; ++i) {
            int k = k0 + ks + i;
            int c = k / 9, r = k - 9 * c;
            int dy = r / 3, dx = r - 3 * dy;
            int yy = y + dy - 1, xx = x + dx - 1;
            __bf16 v = (__bf16)0.f;
            if ((unsigned)yy < 32u && (unsigned)xx < 32u)
                v = fbase[((size_t)c << 10) + (yy << 5) + xx];
            As[buf][m * 40 + ks + i] = v;
        }
    };
    auto loadB = [&](int buf, int k0) {
        if (USE_TDM) {
            if (w == 0)   // 256 rows x 32 bf16, LDS pitch 40 bf16 (pad 4 DW after 16 DW)
                tdm_load_2d(Wb + (size_t)n0 * 1152 + k0, lds_addr_of(&Bs[buf][0]),
                            1152u, 512u, 1152u, 32u, 256u, 1u, 3u, 3u);
        } else {
            const uint4* s4 = (const uint4*)(Wb + (size_t)(n0 + t) * 1152 + k0);
            uint4* d4 = (uint4*)&Bs[buf][t * 40];
            d4[0] = s4[0]; d4[1] = s4[1]; d4[2] = s4[2]; d4[3] = s4[3];
        }
    };

    v8f acc[4][4];
#pragma unroll
    for (int i = 0; i < 4; ++i)
#pragma unroll
        for (int j = 0; j < 4; ++j)
#pragma unroll
            for (int e = 0; e < 8; ++e) acc[i][j][e] = 0.f;

    gatherA(0, 0);
    loadB(0, 0);
    for (int k0 = 0; k0 < 1152; k0 += 32) {
        int cur = (k0 >> 5) & 1, nxt = cur ^ 1;
        if (USE_TDM && w == 0) tdm_join();
        __syncthreads();                          // tile 'cur' ready; 'nxt' free
        if (k0 + 32 < 1152) {                     // overlap next-tile DMA with WMMA
            loadB(nxt, k0 + 32);
            gatherA(nxt, k0 + 32);
        }
        v16bf a[4], bb[4];
#pragma unroll
        for (int tm = 0; tm < 4; ++tm) {          // A frag: K = e+8h | e+8+8h
            int m = 64 * wm + 16 * tm + l15;
            v8bf lo = *(const v8bf*)&As[cur][m * 40 + 8 * h];
            v8bf hi = *(const v8bf*)&As[cur][m * 40 + 16 + 8 * h];
#pragma unroll
            for (int e = 0; e < 8; ++e) { a[tm][e] = lo[e]; a[tm][8 + e] = hi[e]; }
        }
#pragma unroll
        for (int tn = 0; tn < 4; ++tn) {          // B frag: K = e+8h | e+8+8h
            int n = 64 * wn + 16 * tn + l15;
            v8bf lo = *(const v8bf*)&Bs[cur][n * 40 + 8 * h];
            v8bf hi = *(const v8bf*)&Bs[cur][n * 40 + 16 + 8 * h];
#pragma unroll
            for (int e = 0; e < 8; ++e) { bb[tn][e] = lo[e]; bb[tn][8 + e] = hi[e]; }
        }
#pragma unroll
        for (int tm = 0; tm < 4; ++tm)
#pragma unroll
            for (int tn = 0; tn < 4; ++tn)
                acc[tm][tn] = __builtin_amdgcn_wmma_f32_16x16x32_bf16(
                    false, a[tm], false, bb[tn], (short)0, acc[tm][tn], false, false);
    }
    // epilogue: exp + norm scale + 2x2 pool (lane-local via D layout)
#pragma unroll
    for (int tn = 0; tn < 4; ++tn) {
        int nAbs = n0 + 64 * wn + 16 * tn + l15;
#pragma unroll
        for (int tmb = 0; tmb < 2; ++tmb)
#pragma unroll
            for (int rp = 0; rp < 4; ++rp) {
                float s = 0.f;
#pragma unroll
                for (int dtm = 0; dtm < 2; ++dtm)
#pragma unroll
                    for (int dr = 0; dr < 2; ++dr) {
                        int tm = tmb + 2 * dtm, r = 2 * rp + dr;
                        int Ml = 64 * wm + 16 * tm + r + 8 * h;
                        float v = acc[tm][tn][r];
                        s += expf((v * invS[Ml] - 1.0f) * INV_SIG2) * normsS[Ml];
                    }
                int px = 8 * tmb + 4 * h + rp;
                int pos = b * 256 + ((y0 >> 1) + wm) * 16 + px;
                PooledB[(size_t)pos * 512 + nAbs] = (__bf16)(0.25f * s);
            }
    }
}

// ---------------------------------------------------------------- stage B (bf16 WMMA)
__global__ void __launch_bounds__(256) k_stageB(
    const __bf16* __restrict__ PooledB, const __bf16* __restrict__ NrmB,
    float* __restrict__ out) {
    __shared__ __bf16 As[2][128 * 40];
    __shared__ __bf16 Bs[2][256 * 40];
    const int t = threadIdx.x, lane = t & 31, w = t >> 5;
    const int h = lane >> 4, l15 = lane & 15;
    const int wm = w >> 2, wn = w & 3;
    const int mt = blockIdx.x, nt = blockIdx.y;
    const int n0 = nt * 256;

    auto loadTiles = [&](int buf, int k0) {
        if (USE_TDM) {
            if (w == 0) {
                tdm_load_2d(PooledB + (size_t)mt * 128 * 512 + k0, lds_addr_of(&As[buf][0]),
                            512u, 16384u, 512u, 32u, 128u, 1u, 3u, 3u);
                tdm_load_2d(NrmB + (size_t)n0 * 512 + k0, lds_addr_of(&Bs[buf][0]),
                            512u, 512u, 512u, 32u, 256u, 1u, 3u, 3u);
            }
        } else {
            {
                int m = t >> 1, ks = (t & 1) * 16;
                const uint4* s4 = (const uint4*)(PooledB + (size_t)(mt * 128 + m) * 512 + k0 + ks);
                uint4* d4 = (uint4*)&As[buf][m * 40 + ks];
                d4[0] = s4[0]; d4[1] = s4[1];
            }
            {
                const uint4* s4 = (const uint4*)(NrmB + (size_t)(n0 + t) * 512 + k0);
                uint4* d4 = (uint4*)&Bs[buf][t * 40];
                d4[0] = s4[0]; d4[1] = s4[1]; d4[2] = s4[2]; d4[3] = s4[3];
            }
        }
    };

    v8f acc[4][4];
#pragma unroll
    for (int i = 0; i < 4; ++i)
#pragma unroll
        for (int j = 0; j < 4; ++j)
#pragma unroll
            for (int e = 0; e < 8; ++e) acc[i][j][e] = 0.f;

    loadTiles(0, 0);
    for (int k0 = 0; k0 < 512; k0 += 32) {
        int cur = (k0 >> 5) & 1, nxt = cur ^ 1;
        if (USE_TDM && w == 0) tdm_join();
        __syncthreads();
        if (k0 + 32 < 512) loadTiles(nxt, k0 + 32);
        v16bf a[4], bb[4];
#pragma unroll
        for (int tm = 0; tm < 4; ++tm) {
            int m = 64 * wm + 16 * tm + l15;
            v8bf lo = *(const v8bf*)&As[cur][m * 40 + 8 * h];
            v8bf hi = *(const v8bf*)&As[cur][m * 40 + 16 + 8 * h];
#pragma unroll
            for (int e = 0; e < 8; ++e) { a[tm][e] = lo[e]; a[tm][8 + e] = hi[e]; }
        }
#pragma unroll
        for (int tn = 0; tn < 4; ++tn) {
            int n = 64 * wn + 16 * tn + l15;
            v8bf lo = *(const v8bf*)&Bs[cur][n * 40 + 8 * h];
            v8bf hi = *(const v8bf*)&Bs[cur][n * 40 + 16 + 8 * h];
#pragma unroll
            for (int e = 0; e < 8; ++e) { bb[tn][e] = lo[e]; bb[tn][8 + e] = hi[e]; }
        }
#pragma unroll
        for (int tm = 0; tm < 4; ++tm)
#pragma unroll
            for (int tn = 0; tn < 4; ++tn)
                acc[tm][tn] = __builtin_amdgcn_wmma_f32_16x16x32_bf16(
                    false, a[tm], false, bb[tn], (short)0, acc[tm][tn], false, false);
    }
#pragma unroll
    for (int tm = 0; tm < 4; ++tm) {
        int rowBase = mt * 128 + 64 * wm + 16 * tm + 8 * h;   // r = 0
        int bImg = rowBase >> 8, p0 = rowBase & 255;
#pragma unroll
        for (int tn = 0; tn < 4; ++tn) {
            int nAbs = n0 + 64 * wn + 16 * tn + l15;
            float* dst = out + (size_t)bImg * 131072 + (size_t)nAbs * 256 + p0;
            float4 v0 = {acc[tm][tn][0], acc[tm][tn][1], acc[tm][tn][2], acc[tm][tn][3]};
            float4 v1 = {acc[tm][tn][4], acc[tm][tn][5], acc[tm][tn][6], acc[tm][tn][7]};
            *(float4*)dst = v0;
            *(float4*)(dst + 4) = v1;
        }
    }
}

// ---------------------------------------------------------------- launch
extern "C" void kernel_launch(void* const* d_in, const int* in_sizes, int n_in,
                              void* d_out, int out_size, void* d_ws, size_t ws_size,
                              hipStream_t stream) {
    const float* f  = (const float*)d_in[0];   // [64,128,32,32]
    const float* Wf = (const float*)d_in[1];   // [512,1152]
    float* out = (float*)d_out;                // [64,512,16,16]

    char* ws = (char*)d_ws;
    size_t off = 0;
    auto alloc = [&](size_t sz) { char* p = ws + off; off = (off + sz + 255) & ~(size_t)255; return p; };

    __bf16* fb      = (__bf16*)alloc((size_t)64 * 128 * 1024 * 2);
    __bf16* Wb      = (__bf16*)alloc((size_t)512 * 1152 * 2);
    float*  s2      = (float*)alloc((size_t)64 * 1024 * 4);
    float*  norms   = (float*)alloc((size_t)65536 * 4);
    float*  gram    = (float*)alloc((size_t)512 * 512 * 4);
    float*  Y0      = (float*)alloc((size_t)512 * 512 * 4);
    float*  Y1      = (float*)alloc((size_t)512 * 512 * 4);
    float*  Z0      = (float*)alloc((size_t)512 * 512 * 4);
    float*  Z1      = (float*)alloc((size_t)512 * 512 * 4);
    float*  T       = (float*)alloc((size_t)512 * 512 * 4);
    __bf16* NrmB    = (__bf16*)alloc((size_t)512 * 512 * 2);
    __bf16* PooledB = (__bf16*)alloc((size_t)16384 * 512 * 2);
    float*  scal    = (float*)alloc(256);
    float*  sumsq = scal;
    int*    cnt   = (int*)(scal + 1);
    int*    gen   = (int*)(scal + 2);

    k_prep_f<<<256, 256, 0, stream>>>(f, fb, s2);
    k_norms<<<256, 256, 0, stream>>>(s2, norms);
    k_prep_w<<<2304, 256, 0, stream>>>(Wf, Wb, sumsq, cnt, gen);
    k_newton<<<16, 256, 0, stream>>>(Wf, gram, Y0, Y1, Z0, Z1, T, NrmB, sumsq, cnt, gen);
    k_stageA<<<dim3(512, 2), 256, 0, stream>>>(fb, Wb, norms, PooledB);
    k_stageB<<<dim3(128, 2), 256, 0, stream>>>(PooledB, NrmB, out);
}
